// SparseKnowledgeAttention_35553739276536
// MI455X (gfx1250) — matrix operand order repaired and checked by reading, not compiled, not run on hardware
//
#include <hip/hip_runtime.h>
#include <hip/hip_bf16.h>

typedef unsigned short ushort_t;
typedef __attribute__((ext_vector_type(16))) __bf16        v16bf;
typedef __attribute__((ext_vector_type(8)))  float         v8f;
typedef __attribute__((ext_vector_type(4)))  unsigned int  u32x4;
typedef __attribute__((ext_vector_type(4)))  unsigned short u16x4;
typedef __attribute__((ext_vector_type(4)))  float         f32x4;

#define EMB_DIM 256
#define N_EGO   8192
#define N_SIDE  8192
#define TOPK    16
#define NEG_BIG (-3.0e38f)

union Frag32B { u32x4 q[2]; v16bf v; };

// float -> bf16, round-to-nearest-even
__device__ __forceinline__ unsigned short f2bf(float x) {
    unsigned int u = __float_as_uint(x);
    unsigned int r = u + 0x7FFFu + ((u >> 16) & 1u);
    return (unsigned short)(r >> 16);
}

// ---------------------------------------------------------------------------
// Kernel 1: convert operands to bf16 (ego, side*rel, side, Wq, Wk, Wv)
// ---------------------------------------------------------------------------
__global__ void cvt_kernel(const float* __restrict__ ego,
                           const float* __restrict__ side,
                           const float* __restrict__ rel,
                           const float* __restrict__ Wq,
                           const float* __restrict__ Wk,
                           const float* __restrict__ Wv,
                           ushort_t* __restrict__ ego_bf,
                           ushort_t* __restrict__ kin_bf,
                           ushort_t* __restrict__ side_bf,
                           ushort_t* __restrict__ wq_bf,
                           ushort_t* __restrict__ wk_bf,
                           ushort_t* __restrict__ wv_bf) {
    int i = (blockIdx.x * blockDim.x + threadIdx.x) * 4;
    if (i < N_EGO * EMB_DIM) {
        f32x4 e = *(const f32x4*)(ego + i);
        f32x4 s = *(const f32x4*)(side + i);
        f32x4 r = *(const f32x4*)(rel + i);
        u16x4 oe, ok, os;
        oe.x = f2bf(e.x); oe.y = f2bf(e.y); oe.z = f2bf(e.z); oe.w = f2bf(e.w);
        ok.x = f2bf(s.x * r.x); ok.y = f2bf(s.y * r.y);
        ok.z = f2bf(s.z * r.z); ok.w = f2bf(s.w * r.w);
        os.x = f2bf(s.x); os.y = f2bf(s.y); os.z = f2bf(s.z); os.w = f2bf(s.w);
        *(u16x4*)(ego_bf + i)  = oe;
        *(u16x4*)(kin_bf + i)  = ok;
        *(u16x4*)(side_bf + i) = os;
    }
    if (i < EMB_DIM * EMB_DIM) {
        f32x4 a = *(const f32x4*)(Wq + i);
        f32x4 b = *(const f32x4*)(Wk + i);
        f32x4 c = *(const f32x4*)(Wv + i);
        u16x4 oa, ob, oc;
        oa.x = f2bf(a.x); oa.y = f2bf(a.y); oa.z = f2bf(a.z); oa.w = f2bf(a.w);
        ob.x = f2bf(b.x); ob.y = f2bf(b.y); ob.z = f2bf(b.z); ob.w = f2bf(b.w);
        oc.x = f2bf(c.x); oc.y = f2bf(c.y); oc.z = f2bf(c.z); oc.w = f2bf(c.w);
        *(u16x4*)(wq_bf + i) = oa;
        *(u16x4*)(wk_bf + i) = ob;
        *(u16x4*)(wv_bf + i) = oc;
    }
}

// ---------------------------------------------------------------------------
// Kernel 2: projections  out = X @ W.T + b   (NT GEMM, K = 256)
//   blockIdx.y: 0 -> q (bf16 out, scaled by 1/16), 1 -> k (bf16), 2 -> v (f32)
//   One 16x16 output tile per wave; 8 chained v_wmma_f32_16x16x32_bf16.
// ---------------------------------------------------------------------------
__global__ void proj_kernel(const ushort_t* __restrict__ ego_bf,
                            const ushort_t* __restrict__ kin_bf,
                            const ushort_t* __restrict__ side_bf,
                            const ushort_t* __restrict__ wq_bf,
                            const ushort_t* __restrict__ wk_bf,
                            const ushort_t* __restrict__ wv_bf,
                            const float* __restrict__ bq,
                            const float* __restrict__ bk,
                            const float* __restrict__ bv,
                            ushort_t* __restrict__ q_bf,
                            ushort_t* __restrict__ k_bf,
                            float* __restrict__ v_f32) {
    const int mode = blockIdx.y;
    const ushort_t* X; const ushort_t* W; const float* bias;
    if (mode == 0)      { X = ego_bf;  W = wq_bf; bias = bq; }
    else if (mode == 1) { X = kin_bf;  W = wk_bf; bias = bk; }
    else                { X = side_bf; W = wv_bf; bias = bv; }

    const int wave = threadIdx.x >> 5;
    const int lane = threadIdx.x & 31;
    const int half = lane >> 4;
    const int l16  = lane & 15;
    const int tile = blockIdx.x * 8 + wave;          // 8192 tiles: 512 M x 16 N
    const int m0   = (tile >> 4) << 4;
    const int n0   = (tile & 15) << 4;

    const ushort_t* arow = X + (m0 + l16) * EMB_DIM; // A: M = l16
    const ushort_t* brow = W + (n0 + l16) * EMB_DIM; // B column n = W row n

    v8f c = {};
#pragma unroll
    for (int kf = 0; kf < 8; ++kf) {
        const int kb = kf * 32;
        Frag32B a, b;
        a.q[0] = *(const u32x4*)(arow + kb + half * 16);       // K: 16 contig
        a.q[1] = *(const u32x4*)(arow + kb + half * 16 + 8);
        b.q[0] = *(const u32x4*)(brow + kb + half * 16);       // K: 16 contig
        b.q[1] = *(const u32x4*)(brow + kb + half * 16 + 8);
        c = __builtin_amdgcn_wmma_f32_16x16x32_bf16(false, a.v, false, b.v,
                                                    (short)0, c, false, false);
    }

    const int n = n0 + l16;
    const float bv_ = bias[n];
#pragma unroll
    for (int r = 0; r < 8; ++r) {
        const int m = m0 + r + half * 8;
        const float val = c[r] + bv_;
        if (mode == 0)      q_bf[m * EMB_DIM + n] = f2bf(val * 0.0625f); // fold 1/sqrt(256)
        else if (mode == 1) k_bf[m * EMB_DIM + n] = f2bf(val);
        else                v_f32[m * EMB_DIM + n] = val;
    }
}

// ---------------------------------------------------------------------------
// Top-k register insertion: unsorted list + running min; rare replace.
// ---------------------------------------------------------------------------
__device__ __forceinline__ void topk_insert(float s, int sidx,
                                            float (&tv)[TOPK], int (&ti)[TOPK],
                                            float& mn, int& mp) {
    if (s > mn) {
#pragma unroll
        for (int j = 0; j < TOPK; ++j)
            if (j == mp) { tv[j] = s; ti[j] = sidx; }
        mn = tv[0]; mp = 0;
#pragma unroll
        for (int j = 1; j < TOPK; ++j)
            if (tv[j] < mn) { mn = tv[j]; mp = j; }
    }
}

// ---------------------------------------------------------------------------
// Kernel 3: fused scores (k @ q^T via WMMA) + streaming top-16 + softmax +
// weighted gather of v.  One wave = 16 ego rows; 256-thread block = 128 rows.
// Score tile layout: lane holds ego col N = lane%16, side rows M = r+8*(lane/16)
// -> each ego row's candidates live in exactly 2 lanes; merge via shfl_xor(16).
// ---------------------------------------------------------------------------
__global__ void attn_kernel(const ushort_t* __restrict__ q_bf,
                            const ushort_t* __restrict__ k_bf,
                            const float* __restrict__ v_f32,
                            float* __restrict__ out) {
    __shared__ float sw[8][16][TOPK];
    __shared__ int   si[8][16][TOPK];

    const int wave = threadIdx.x >> 5;
    const int lane = threadIdx.x & 31;
    const int half = lane >> 4;
    const int l16  = lane & 15;
    const int e0   = blockIdx.x * 128 + wave * 16;

    // B fragments (q rows of this ego tile), resident for the whole stream
    Frag32B B[8];
    const ushort_t* qrow = q_bf + (e0 + l16) * EMB_DIM + half * 16;
#pragma unroll
    for (int kf = 0; kf < 8; ++kf) {
        B[kf].q[0] = *(const u32x4*)(qrow + kf * 32);
        B[kf].q[1] = *(const u32x4*)(qrow + kf * 32 + 8);
    }

    float tv[TOPK]; int ti[TOPK];
#pragma unroll
    for (int j = 0; j < TOPK; ++j) { tv[j] = NEG_BIG; ti[j] = 0; }
    float mn = NEG_BIG; int mp = 0;

    for (int s0 = 0; s0 < N_SIDE; s0 += 16) {
        const ushort_t* krow = k_bf + (s0 + l16) * EMB_DIM;
        v8f c = {};
#pragma unroll
        for (int kf = 0; kf < 8; ++kf) {
            const int kb = kf * 32;
            Frag32B a;
            a.q[0] = *(const u32x4*)(krow + kb + half * 16);
            a.q[1] = *(const u32x4*)(krow + kb + half * 16 + 8);
            c = __builtin_amdgcn_wmma_f32_16x16x32_bf16(false, a.v, false, B[kf].v,
                                                        (short)0, c, false, false);
        }
        const int sbase = s0 + half * 8;
#pragma unroll
        for (int r = 0; r < 8; ++r)
            topk_insert(c[r], sbase + r, tv, ti, mn, mp);
    }

    // merge the two half-wave partial top-16 lists (snapshot first!)
    float ov[TOPK]; int oi[TOPK];
#pragma unroll
    for (int j = 0; j < TOPK; ++j) {
        ov[j] = __shfl_xor(tv[j], 16);
        oi[j] = __shfl_xor(ti[j], 16);
    }
#pragma unroll
    for (int j = 0; j < TOPK; ++j)
        topk_insert(ov[j], oi[j], tv, ti, mn, mp);

    // softmax over the 16 winners (lanes 0..15 hold final per-row result)
    float mx = tv[0];
#pragma unroll
    for (int j = 1; j < TOPK; ++j) mx = fmaxf(mx, tv[j]);
    float wts[TOPK]; float sum = 0.f;
#pragma unroll
    for (int j = 0; j < TOPK; ++j) { wts[j] = __expf(tv[j] - mx); sum += wts[j]; }
    const float inv = 1.0f / sum;

    if (half == 0) {
#pragma unroll
        for (int j = 0; j < TOPK; ++j) {
            sw[wave][l16][j] = wts[j] * inv;
            si[wave][l16][j] = ti[j];
        }
    }
    __syncthreads();

    // aggregation: whole wave cooperates per ego row; coalesced v reads
    for (int r = 0; r < 16; ++r) {
        float w[TOPK]; const float* vrow[TOPK];
#pragma unroll
        for (int i = 0; i < TOPK; ++i) {
            w[i]    = sw[wave][r][i];
            vrow[i] = v_f32 + si[wave][r][i] * EMB_DIM;
        }
        float* orow = out + (e0 + r) * EMB_DIM;
#pragma unroll
        for (int ch = 0; ch < 8; ++ch) {
            const int d = ch * 32 + lane;
            float acc = 0.f;
#pragma unroll
            for (int i = 0; i < TOPK; ++i) acc += w[i] * vrow[i][d];
            orow[d] = acc;
        }
    }
}

// ---------------------------------------------------------------------------
extern "C" void kernel_launch(void* const* d_in, const int* in_sizes, int n_in,
                              void* d_out, int out_size, void* d_ws, size_t ws_size,
                              hipStream_t stream) {
    const float* ego  = (const float*)d_in[0];
    const float* side = (const float*)d_in[1];
    const float* rel  = (const float*)d_in[2];
    const float* Wq   = (const float*)d_in[3];
    const float* bq   = (const float*)d_in[4];
    const float* Wk   = (const float*)d_in[5];
    const float* bk   = (const float*)d_in[6];
    const float* Wv   = (const float*)d_in[7];
    const float* bv   = (const float*)d_in[8];
    float* out = (float*)d_out;

    char* ws = (char*)d_ws;
    const size_t MB = 1024 * 1024;
    ushort_t* ego_bf  = (ushort_t*)(ws + 0 * MB);     // 4 MiB
    ushort_t* kin_bf  = (ushort_t*)(ws + 4 * MB);     // 4 MiB
    ushort_t* side_bf = (ushort_t*)(ws + 8 * MB);     // 4 MiB
    ushort_t* q_bf    = (ushort_t*)(ws + 12 * MB);    // 4 MiB
    ushort_t* k_bf    = (ushort_t*)(ws + 16 * MB);    // 4 MiB
    float*    v_f32   = (float*)   (ws + 20 * MB);    // 8 MiB
    ushort_t* wq_bf   = (ushort_t*)(ws + 28 * MB);            // 128 KiB
    ushort_t* wk_bf   = (ushort_t*)(ws + 28 * MB + 131072);   // 128 KiB
    ushort_t* wv_bf   = (ushort_t*)(ws + 28 * MB + 262144);   // 128 KiB

    // 1) bf16 conversions (2M elems / 4 per thread)
    cvt_kernel<<<2048, 256, 0, stream>>>(ego, side, rel, Wq, Wk, Wv,
                                         ego_bf, kin_bf, side_bf,
                                         wq_bf, wk_bf, wv_bf);

    // 2) q/k/v projections: 8192 16x16 tiles per projection, 8 waves per block
    proj_kernel<<<dim3(1024, 3), 256, 0, stream>>>(ego_bf, kin_bf, side_bf,
                                                   wq_bf, wk_bf, wv_bf,
                                                   bq, bk, bv,
                                                   q_bf, k_bf, v_f32);

    // 3) fused scores + top-16 + softmax + aggregate: 64 blocks x 128 ego rows
    attn_kernel<<<64, 256, 0, stream>>>(q_bf, k_bf, v_f32, out);
}
